// VQVAE_34643206210158
// MI455X (gfx1250) — compile-verified, matrix operand order checked by reading, not run on hardware
//
#include <hip/hip_runtime.h>

typedef __attribute__((ext_vector_type(16))) _Float16 v16h;
typedef __attribute__((ext_vector_type(8)))  float    v8f;

// ---------------------------------------------------------------------------
// Fragment load: 16 f16 along K for one row (A) / column (B), as two 16B chunks
// ---------------------------------------------------------------------------
__device__ inline v16h ldfrag(const _Float16* p) {
  union { uint4 q[2]; v16h v; } u;
  u.q[0] = *(const uint4*)(p);
  u.q[1] = *(const uint4*)(p + 16);
  return u.v;
}

// ---------------------------------------------------------------------------
// Unified WMMA GEMM:  C[M,N] = A[Mpad,Kpad] * B[N,Kpad]^T  (+bias, relu, or VQ)
// mode: 0 = bias+store, 1 = bias+relu+store, 2 = VQ argmin reduction
// Double-buffered LDS; global loads for tile k+1 are issued before the WMMAs
// of tile k so HBM/L2 latency overlaps matrix math. One barrier per K-tile.
// Output addressing: out[m*sM + b*sB + prow*rowMul + pcol*colMul + outOff]
// where n = (b << pShift) | p, p = (prow << owShift) | pcol
// ---------------------------------------------------------------------------
__global__ __launch_bounds__(256) void gemm_wmma_kernel(
    const _Float16* __restrict__ A, const _Float16* __restrict__ B,
    const float* __restrict__ bias, float* __restrict__ out,
    unsigned long long* __restrict__ vqmin, const float* __restrict__ norms,
    int M, int Kpad, int N,
    int sM, int sB, int pShift, int rowMul, int colMul, int owShift, int outOff,
    int mode)
{
  __shared__ __align__(16) _Float16 As[2][128 * 40];
  __shared__ __align__(16) _Float16 Bs[2][128 * 40];

  const int t     = threadIdx.x;
  const int lane  = t & 31;
  const int wid   = t >> 5;
  const int wm    = (wid & 3) * 32;   // wave M offset within block tile
  const int wn    = (wid >> 2) * 64;  // wave N offset within block tile
  const int mBase = blockIdx.y * 128;
  const int nBase = blockIdx.x * 128;
  const int l16   = lane & 15;
  const int hi8   = (lane >> 4) * 8;  // K-chunk base (halves) AND C-frag M offset

  // each thread stages two 16B chunks of A-tile and two of B-tile
  const int r0 = t >> 2,          c0 = (t & 3) * 8;          // chunk 0
  const int r1 = (t + 256) >> 2,  c1 = ((t + 256) & 3) * 8;  // chunk 1
  const _Float16* Ap = A + (size_t)mBase * Kpad;
  const _Float16* Bp = B + (size_t)nBase * Kpad;

  v8f acc[2][4] = {};

  // prologue: stage K-tile 0 into buffer 0
  {
    uint4 a0 = *(const uint4*)(Ap + (size_t)r0 * Kpad + c0);
    uint4 a1 = *(const uint4*)(Ap + (size_t)r1 * Kpad + c1);
    uint4 b0 = *(const uint4*)(Bp + (size_t)r0 * Kpad + c0);
    uint4 b1 = *(const uint4*)(Bp + (size_t)r1 * Kpad + c1);
    *(uint4*)&As[0][r0 * 40 + c0] = a0;
    *(uint4*)&As[0][r1 * 40 + c1] = a1;
    *(uint4*)&Bs[0][r0 * 40 + c0] = b0;
    *(uint4*)&Bs[0][r1 * 40 + c1] = b1;
  }
  __syncthreads();

  const int nIter = Kpad >> 5;
  for (int it = 0; it < nIter; ++it) {
    const int cur = it & 1;
    const int nxt = cur ^ 1;
    const int kn  = (it + 1) << 5;
    const bool has = kn < Kpad;

    // issue next tile's global loads first (latency overlaps the WMMAs below)
    uint4 a0, a1, b0, b1;
    if (has) {
      a0 = *(const uint4*)(Ap + (size_t)r0 * Kpad + kn + c0);
      a1 = *(const uint4*)(Ap + (size_t)r1 * Kpad + kn + c1);
      b0 = *(const uint4*)(Bp + (size_t)r0 * Kpad + kn + c0);
      b1 = *(const uint4*)(Bp + (size_t)r1 * Kpad + kn + c1);
      if (kn + 32 < Kpad) {   // pre-stage L2 two tiles ahead
        __builtin_prefetch(Ap + (size_t)r0 * Kpad + kn + 32 + c0, 0, 1);
        __builtin_prefetch(Bp + (size_t)r0 * Kpad + kn + 32 + c0, 0, 1);
      }
    }

    v16h af[2], bf[4];
#pragma unroll
    for (int i = 0; i < 2; ++i)
      af[i] = ldfrag(&As[cur][(wm + i * 16 + l16) * 40 + hi8]);
#pragma unroll
    for (int j = 0; j < 4; ++j)
      bf[j] = ldfrag(&Bs[cur][(wn + j * 16 + l16) * 40 + hi8]);

#pragma unroll
    for (int i = 0; i < 2; ++i)
#pragma unroll
      for (int j = 0; j < 4; ++j)
        acc[i][j] = __builtin_amdgcn_wmma_f32_16x16x32_f16(
            false, af[i], false, bf[j], (short)0, acc[i][j], false, false);

    if (has) {
      *(uint4*)&As[nxt][r0 * 40 + c0] = a0;
      *(uint4*)&As[nxt][r1 * 40 + c1] = a1;
      *(uint4*)&Bs[nxt][r0 * 40 + c0] = b0;
      *(uint4*)&Bs[nxt][r1 * 40 + c1] = b1;
    }
    __syncthreads();
  }

  if (mode == 2) {
    // VQ: dist(m,n) = ||e_m||^2 - 2 * (e_m . z_n); argmin over m per column n
#pragma unroll
    for (int j = 0; j < 4; ++j) {
      float bd = 3.0e38f; int bm = 0;
#pragma unroll
      for (int i = 0; i < 2; ++i)
#pragma unroll
        for (int r = 0; r < 8; ++r) {
          int m = mBase + wm + i * 16 + r + hi8;
          float d = norms[m] - 2.0f * acc[i][j][r];
          if (d < bd) { bd = d; bm = m; }
        }
      float od = __shfl_xor(bd, 16, 32);   // lanes l and l+16 share column N=l
      int   om = __shfl_xor(bm, 16, 32);
      if (od < bd || (od == bd && om < bm)) { bd = od; bm = om; }
      if (lane < 16) {
        int n = nBase + wn + j * 16 + l16;
        unsigned u = __float_as_uint(bd);
        u = (u & 0x80000000u) ? ~u : (u | 0x80000000u);  // sortable float key
        unsigned long long key =
            ((unsigned long long)u << 32) | (unsigned long long)(unsigned)bm;
        atomicMin(vqmin + n, key);
      }
    }
    return;
  }

  const int pMask  = (1 << pShift) - 1;
  const int owMask = (1 << owShift) - 1;
#pragma unroll
  for (int j = 0; j < 4; ++j) {
    int n    = nBase + wn + j * 16 + l16;
    int bimg = n >> pShift;
    int p    = n & pMask;
    size_t base = (size_t)bimg * sB + (size_t)(p >> owShift) * rowMul
                + (size_t)(p & owMask) * colMul + (size_t)outOff;
#pragma unroll
    for (int i = 0; i < 2; ++i)
#pragma unroll
      for (int r = 0; r < 8; ++r) {
        int m = mBase + wm + i * 16 + r + hi8;
        if (m < M) {
          float v = acc[i][j][r];
          if (bias) v += bias[m];
          if (mode == 1) v = fmaxf(v, 0.0f);
          out[(size_t)m * sM + base] = v;
        }
      }
  }
}

// ---------------------------------------------------------------------------
// im2col: src f32 (strided NCHW or C-major) -> B matrix f16 [N][Kpad]
// k = c*kh*kw + ky*kw + kx ; n = (b<<pShift)|p ; p = (oy<<owShift)|ox
// ---------------------------------------------------------------------------
__global__ void im2col_kernel(const float* __restrict__ src,
                              _Float16* __restrict__ dst,
                              long long total, int Kpad, int Kreal,
                              int H, int W, int pShift, int owShift,
                              int kh, int kw, int stride, int padY, int padX,
                              int lhsDil, long long srcStrideB, long long srcStrideC)
{
  long long gstride = (long long)gridDim.x * blockDim.x;
  for (long long idx = (long long)blockIdx.x * blockDim.x + threadIdx.x;
       idx < total; idx += gstride) {
    int k = (int)(idx % Kpad);
    long long n = idx / Kpad;
    float v = 0.0f;
    if (k < Kreal) {
      int kx = k % kw; int tt = k / kw; int ky = tt % kh; int c = tt / kh;
      int bimg = (int)(n >> pShift);
      int p    = (int)n & ((1 << pShift) - 1);
      int oy = p >> owShift, ox = p & ((1 << owShift) - 1);
      int iy = oy * stride - padY + ky;
      int ix = ox * stride - padX + kx;
      bool ok = true;
      if (lhsDil == 2) { ok = ((iy | ix) & 1) == 0; iy >>= 1; ix >>= 1; }
      ok = ok && iy >= 0 && iy < H && ix >= 0 && ix < W;
      if (ok) v = src[(long long)bimg * srcStrideB + (long long)c * srcStrideC
                      + (long long)iy * W + ix];
    }
    dst[idx] = (_Float16)v;
  }
}

// ---------------------------------------------------------------------------
// Weight pack: OIHW f32 -> [Mpad][Kpad] f16, with strided kernel-tap selection
// (startY/stepY/cntY etc.) for transposed-conv parity sub-kernels.
// ---------------------------------------------------------------------------
__global__ void packw_kernel(const float* __restrict__ w, _Float16* __restrict__ dst,
                             int M, int Mpad, int I, int kh, int kw,
                             int startY, int stepY, int cntY,
                             int startX, int stepX, int cntX, int Kpad)
{
  int Kreal = I * cntY * cntX;
  long long total = (long long)Mpad * Kpad;
  long long gstride = (long long)gridDim.x * blockDim.x;
  for (long long idx = (long long)blockIdx.x * blockDim.x + threadIdx.x;
       idx < total; idx += gstride) {
    int k = (int)(idx % Kpad);
    int m = (int)(idx / Kpad);
    float v = 0.0f;
    if (m < M && k < Kreal) {
      int kx = k % cntX; int tt = k / cntX; int ky = tt % cntY; int c = tt / cntY;
      v = w[((long long)m * I + c) * (kh * kw)
            + (startY + ky * stepY) * kw + (startX + kx * stepX)];
    }
    dst[idx] = (_Float16)v;
  }
}

__global__ void code_norms_kernel(const float* __restrict__ cb,
                                  float* __restrict__ norms, int K, int D) {
  int k = blockIdx.x * blockDim.x + threadIdx.x;
  if (k < K) {
    float s = 0.0f;
    for (int d = 0; d < D; ++d) { float e = cb[(long long)k * D + d]; s += e * e; }
    norms[k] = s;
  }
}

__global__ void vq_init_kernel(unsigned long long* v, int n) {
  int i = blockIdx.x * blockDim.x + threadIdx.x;
  if (i < n) v[i] = ~0ULL;
}

// extract codes; gather z_q in C-major [D][N] f32 layout for the decoder
__global__ void vq_select_kernel(const unsigned long long* __restrict__ vqmin,
                                 const float* __restrict__ cb,
                                 float* __restrict__ zq, int* __restrict__ codes,
                                 int N, int D) {
  long long total = (long long)N * D;
  long long gstride = (long long)gridDim.x * blockDim.x;
  for (long long idx = (long long)blockIdx.x * blockDim.x + threadIdx.x;
       idx < total; idx += gstride) {
    int n = (int)(idx % N);
    int d = (int)(idx / N);
    int code = (int)(vqmin[n] & 0xFFFFFFFFULL);
    zq[idx] = cb[(long long)code * D + d];
    if (d == 0) codes[n] = code;
  }
}

// ---------------------------------------------------------------------------
extern "C" void kernel_launch(void* const* d_in, const int* in_sizes, int n_in,
                              void* d_out, int out_size, void* d_ws, size_t ws_size,
                              hipStream_t stream) {
  (void)in_sizes; (void)n_in; (void)out_size; (void)ws_size;
  const float* x      = (const float*)d_in[0];
  const float* cb     = (const float*)d_in[1];
  const float* enc_w1 = (const float*)d_in[2];
  const float* enc_b1 = (const float*)d_in[3];
  const float* enc_w2 = (const float*)d_in[4];
  const float* enc_b2 = (const float*)d_in[5];
  const float* enc_w3 = (const float*)d_in[6];
  const float* enc_b3 = (const float*)d_in[7];
  const float* dec_w1 = (const float*)d_in[8];
  const float* dec_b1 = (const float*)d_in[9];
  const float* dec_w2 = (const float*)d_in[10];
  const float* dec_b2 = (const float*)d_in[11];
  const float* dec_w3 = (const float*)d_in[12];
  const float* dec_b3 = (const float*)d_in[13];

  float* xrec = (float*)d_out;
  int*   codes = (int*)((float*)d_out + (size_t)32 * 3 * 128 * 128);

  char* ws = (char*)d_ws;
  size_t off = 0;
  auto alloc = [&](size_t bytes) -> void* {
    void* p = ws + off;
    off += (bytes + 255) & ~(size_t)255;
    return p;
  };
  _Float16* wA                 = (_Float16*)alloc((size_t)2 << 20);             // packed weights (max 1.125MB)
  float* norms                 = (float*)alloc(1024 * sizeof(float));
  unsigned long long* vqmin    = (unsigned long long*)alloc(32768ull * 8);
  _Float16* Bbuf               = (_Float16*)alloc((size_t)2304 * 32768 * 2);    // im2col (max 144MB)
  float* slotA                 = (float*)alloc((size_t)128 * 131072 * 4);       // act1 / h2
  float* slotB                 = (float*)alloc((size_t)256 * 32768 * 4);        // act2 / z_e / h1
  float* slotC                 = (float*)alloc((size_t)256 * 32768 * 4);        // z_q

  auto gemm = [&](const _Float16* Bm, const float* bias, float* out,
                  int M, int Mpad, int Kpad, int N, int sM, int sB, int pShift,
                  int rowMul, int colMul, int owShift, int outOff, int mode) {
    dim3 grid(N / 128, Mpad / 128);
    gemm_wmma_kernel<<<grid, 256, 0, stream>>>(wA, Bm, bias, out, vqmin, norms,
        M, Kpad, N, sM, sB, pShift, rowMul, colMul, owShift, outOff, mode);
  };
  auto im2col = [&](const float* src, int N, int Kpad, int Kreal, int H, int W,
                    int pShift, int owShift, int kh, int kw, int stride,
                    int padY, int padX, int lhsDil, long long sBs, long long sCs) {
    long long total = (long long)N * Kpad;
    int blocks = (int)((total + 255) / 256);
    im2col_kernel<<<blocks, 256, 0, stream>>>(src, Bbuf, total, Kpad, Kreal,
        H, W, pShift, owShift, kh, kw, stride, padY, padX, lhsDil, sBs, sCs);
  };
  auto packw = [&](const float* w, int M, int Mpad, int I, int kh, int kw,
                   int sY, int stY, int cY, int sX, int stX, int cX, int Kpad) {
    long long total = (long long)Mpad * Kpad;
    int blocks = (int)((total + 255) / 256);
    packw_kernel<<<blocks, 256, 0, stream>>>(w, wA, M, Mpad, I, kh, kw,
        sY, stY, cY, sX, stX, cX, Kpad);
  };

  // ---- enc1: x[32,3,128,128] --4x4 s2 p1--> act1 [128][32*64*64], relu
  packw(enc_w1, 128, 128, 3, 4, 4, 0,1,4, 0,1,4, 64);
  im2col(x, 131072, 64, 48, 128, 128, 12, 6, 4, 4, 2, 1, 1, 1,
         (long long)3 * 16384, 16384);
  gemm(Bbuf, enc_b1, slotA, 128, 128, 64, 131072, 131072, 4096, 12, 64, 1, 6, 0, 1);

  // ---- enc2: act1 --4x4 s2 p1--> act2 [256][32*32*32], relu
  packw(enc_w2, 256, 256, 128, 4, 4, 0,1,4, 0,1,4, 2048);
  im2col(slotA, 32768, 2048, 2048, 64, 64, 10, 5, 4, 4, 2, 1, 1, 1, 4096, 131072);
  gemm(Bbuf, enc_b2, slotB, 256, 256, 2048, 32768, 32768, 1024, 10, 32, 1, 5, 0, 1);

  // ---- enc3: act2 --3x3 s1 p1--> z_e (overwrites slotB; GEMM reads only Bbuf)
  packw(enc_w3, 256, 256, 256, 3, 3, 0,1,3, 0,1,3, 2304);
  im2col(slotB, 32768, 2304, 2304, 32, 32, 10, 5, 3, 3, 1, 1, 1, 1, 1024, 32768);
  gemm(Bbuf, enc_b3, slotB, 256, 256, 2304, 32768, 32768, 1024, 10, 32, 1, 5, 0, 0);

  // ---- VQ: scores = codebook * z ; fused argmin via packed u64 atomic min
  packw(cb, 1024, 1024, 256, 1, 1, 0,1,1, 0,1,1, 256);
  code_norms_kernel<<<4, 256, 0, stream>>>(cb, norms, 1024, 256);
  vq_init_kernel<<<128, 256, 0, stream>>>(vqmin, 32768);
  im2col(slotB, 32768, 256, 256, 32, 32, 10, 5, 1, 1, 1, 0, 0, 1, 1024, 32768);
  gemm(Bbuf, nullptr, nullptr, 1024, 1024, 256, 32768, 0, 0, 0, 0, 0, 0, 0, 2);
  vq_select_kernel<<<32768, 256, 0, stream>>>(vqmin, cb, slotC, codes, 32768, 256);

  // ---- dec1: z_q --3x3 s1 p1--> h1 [256][32768], relu
  packw(dec_w1, 256, 256, 256, 3, 3, 0,1,3, 0,1,3, 2304);
  im2col(slotC, 32768, 2304, 2304, 32, 32, 10, 5, 3, 3, 1, 1, 1, 1, 1024, 32768);
  gemm(Bbuf, dec_b1, slotB, 256, 256, 2304, 32768, 32768, 1024, 10, 32, 1, 5, 0, 1);

  // ---- dec2: ConvT(4,2,1) as 4 parity sub-convs: h1 -> h2 [128][32*64*64], relu
  for (int py = 0; py < 2; ++py)
    for (int px = 0; px < 2; ++px) {
      packw(dec_w2, 128, 128, 256, 4, 4, py,2,2, px,2,2, 1024);
      im2col(slotB, 32768, 1024, 1024, 32, 32, 10, 5, 2, 2, 1,
             1 - py, 1 - px, 1, 1024, 32768);
      gemm(Bbuf, dec_b2, slotA, 128, 128, 1024, 32768,
           131072, 4096, 10, 128, 2, 5, py * 64 + px, 1);
    }

  // ---- dec3: ConvT(4,2,1) as 4 parity sub-convs: h2 -> x_recon NCHW [32,3,128,128]
  for (int py = 0; py < 2; ++py)
    for (int px = 0; px < 2; ++px) {
      packw(dec_w3, 3, 128, 128, 4, 4, py,2,2, px,2,2, 512);
      im2col(slotA, 131072, 512, 512, 64, 64, 12, 6, 2, 2, 1,
             1 - py, 1 - px, 1, 4096, 131072);
      gemm(Bbuf, dec_b3, xrec, 3, 128, 512, 131072,
           16384, 49152, 12, 256, 2, 6, py * 128 + px, 0);
    }
}
